// PrefixMultiHeadAttention_30030411333684
// MI455X (gfx1250) — matrix operand order repaired and checked by reading, not compiled, not run on hardware
//
#include <hip/hip_runtime.h>
#include <hip/hip_bf16.h>

// ---------------- problem constants ----------------
constexpr int Bc  = 4;
constexpr int Sc  = 2048;
constexpr int Dc  = 1024;
constexpr int Hc  = 16;
constexpr int Pc  = 10;
constexpr int DKc = 64;
constexpr int Lc  = Pc + Sc;        // 2058 keys
constexpr int Lp  = 2080;           // padded to multiple of 32
constexpr int Mrows = Bc * Sc;      // 8192

typedef __attribute__((ext_vector_type(16))) __bf16 v16bf;
typedef __attribute__((ext_vector_type(8)))  float  v8f;

union AFragU {
    v16bf v;
    ulonglong2 q[2];
    __bf16 e[16];
};

static __device__ inline v8f wmma_bf16(v16bf a, v16bf b, v8f c) {
    // D = A(16x32 bf16) * B(32x16 bf16) + C(16x16 f32)
    return __builtin_amdgcn_wmma_f32_16x16x32_bf16(
        /*neg_a=*/false, a, /*neg_b=*/false, b,
        /*c_mod=*/(short)0, c, /*reuse_a=*/false, /*reuse_b=*/false);
}

static __device__ inline v8f zero_v8f() {
    v8f z;
#pragma unroll
    for (int i = 0; i < 8; ++i) z[i] = 0.0f;
    return z;
}

static __device__ inline void cvt4(__bf16* dst, float4 f) {
    dst[0] = (__bf16)f.x; dst[1] = (__bf16)f.y;
    dst[2] = (__bf16)f.z; dst[3] = (__bf16)f.w;
}

// ---------------- weight transpose + bf16 convert: Wt[n][k] = W[k][n] ----------------
__global__ void wt_conv_kernel(const float* __restrict__ W, __bf16* __restrict__ Wt) {
    int idx = blockIdx.x * blockDim.x + threadIdx.x;   // over D*D
    int n = idx >> 10;          // / Dc
    int k = idx & (Dc - 1);     // % Dc
    Wt[idx] = (__bf16)W[(size_t)k * Dc + n];
}

// ---------------- prefix + pad fill for K (row-major) and V (transposed) ----------------
__global__ void prefix_fill_kernel(const float* __restrict__ pk, const float* __restrict__ pv,
                                   __bf16* __restrict__ Kbf, __bf16* __restrict__ Vt) {
    // rows handled per (b,h): [0,P) prefix + [Lc, Lp) zero pad  => 32 rows, 64 dk each
    int idx = blockIdx.x * blockDim.x + threadIdx.x;   // B*H*32*64 threads
    int dk  = idx & 63;
    int ri  = (idx >> 6) & 31;
    int bh  = idx >> 11;
    if (bh >= Bc * Hc) return;
    int h = bh & (Hc - 1);
    int row = (ri < Pc) ? ri : (Lc + (ri - Pc));
    __bf16 kv = (__bf16)0.0f, vv = (__bf16)0.0f;
    if (ri < Pc) {
        kv = (__bf16)pk[((size_t)h * Pc + ri) * DKc + dk];
        vv = (__bf16)pv[((size_t)h * Pc + ri) * DKc + dk];
    }
    Kbf[((size_t)bh * Lp + row) * DKc + dk] = kv;
    Vt[((size_t)bh * DKc + dk) * Lp + row] = vv;
}

// ---------------- generic projection / output GEMM ----------------
// C[M=8192, N=1024] = A(f32, row major, K=1024) * Bt(bf16, [N][K]) + bias
// MODE 0: f32 out [row][col]          (final projection)
// MODE 1: bf16 out Q  [B,H,S,DK]
// MODE 2: bf16 out K  [B,H,Lp,DK]  at row offset P
// MODE 3: bf16 out Vt [B,H,DK,Lp]  at key offset P
template<int MODE>
__global__ __launch_bounds__(256)
void gemm_kernel(const float* __restrict__ A, const __bf16* __restrict__ Bt,
                 const float* __restrict__ bias, void* __restrict__ out) {
    const int lane = threadIdx.x & 31;
    const int w    = threadIdx.x >> 5;     // 0..7
    const int wm   = w & 3;                // 4 waves along M
    const int wn   = w >> 2;               // 2 waves along N
    const int m0   = lane & 15;
    const int hh   = lane >> 4;            // lane half
    const int khalf = hh * 16;             // A frag K offset
    const int ksec  = hh * 16;             // B frag K offset

    const int rowBase = blockIdx.y * 128 + wm * 32;
    const int colBase = blockIdx.x * 128 + wn * 64;

    v8f acc[2][4];
#pragma unroll
    for (int mi = 0; mi < 2; ++mi)
#pragma unroll
        for (int ni = 0; ni < 4; ++ni) acc[mi][ni] = zero_v8f();

    for (int kb = 0; kb < Dc; kb += 32) {
        AFragU a[2];
#pragma unroll
        for (int mi = 0; mi < 2; ++mi) {
            const int row = rowBase + mi * 16 + m0;
            const float* ap = A + (size_t)row * Dc + kb + khalf;
            float4 f0 = *(const float4*)(ap);
            float4 f1 = *(const float4*)(ap + 4);
            float4 f2 = *(const float4*)(ap + 8);
            float4 f3 = *(const float4*)(ap + 12);
            cvt4(&a[mi].e[0],  f0);
            cvt4(&a[mi].e[4],  f1);
            cvt4(&a[mi].e[8],  f2);
            cvt4(&a[mi].e[12], f3);
            if (kb + 32 < Dc) __builtin_prefetch(ap + 32, 0, 1);
        }
        v16bf b[4];
#pragma unroll
        for (int ni = 0; ni < 4; ++ni) {
            const int col = colBase + ni * 16 + m0;
            const __bf16* bp = Bt + (size_t)col * Dc + kb + ksec;
            b[ni] = *(const v16bf*)bp;
            if (kb + 32 < Dc) __builtin_prefetch(bp + 32, 0, 1);
        }
#pragma unroll
        for (int mi = 0; mi < 2; ++mi)
#pragma unroll
            for (int ni = 0; ni < 4; ++ni)
                acc[mi][ni] = wmma_bf16(a[mi].v, b[ni], acc[mi][ni]);
    }

    // epilogue: bias + scatter per output mode (C layout: VGPR r -> M = r + 8*hh, N = m0)
#pragma unroll
    for (int ni = 0; ni < 4; ++ni) {
        const int col = colBase + ni * 16 + m0;
        const float bv = bias[col];
#pragma unroll
        for (int mi = 0; mi < 2; ++mi) {
#pragma unroll
            for (int r = 0; r < 8; ++r) {
                const int row = rowBase + mi * 16 + r + 8 * hh;
                const float val = acc[mi][ni][r] + bv;
                if (MODE == 0) {
                    ((float*)out)[(size_t)row * Dc + col] = val;
                } else {
                    const int bb = row >> 11;            // / Sc
                    const int s  = row & (Sc - 1);
                    const int h  = col >> 6;             // / DKc
                    const int dk = col & (DKc - 1);
                    __bf16* o = (__bf16*)out;
                    if (MODE == 1)
                        o[(((size_t)bb * Hc + h) * Sc + s) * DKc + dk] = (__bf16)val;
                    else if (MODE == 2)
                        o[(((size_t)bb * Hc + h) * Lp + Pc + s) * DKc + dk] = (__bf16)val;
                    else // MODE == 3
                        o[(((size_t)bb * Hc + h) * DKc + dk) * Lp + Pc + s] = (__bf16)val;
                }
            }
        }
    }
}

// ---------------- flash attention: one wave per 16-query tile ----------------
__global__ __launch_bounds__(256)
void flash_attn_kernel(const __bf16* __restrict__ Q, const __bf16* __restrict__ K,
                       const __bf16* __restrict__ Vt, float* __restrict__ O32) {
    const int bh   = blockIdx.x;                 // B*H
    const int lane = threadIdx.x & 31;
    const int w    = threadIdx.x >> 5;           // wave 0..7
    const int qblock = blockIdx.y * 128;
    const int qbase  = qblock + w * 16;
    const int m0   = lane & 15;
    const int hh   = lane >> 4;
    const int khalf = hh * 16;
    const int ksec  = hh * 16;

    const __bf16* Qp = Q  + ((size_t)bh * Sc + qbase) * DKc;
    const __bf16* Kp = K  + (size_t)bh * Lp * DKc;
    const __bf16* Vp = Vt + (size_t)bh * DKc * Lp;

    // Q A-fragments for dk chunks [0,32) and [32,64)
    AFragU qa[2];
#pragma unroll
    for (int kc = 0; kc < 2; ++kc) {
        const __bf16* p = Qp + (size_t)m0 * DKc + kc * 32 + khalf;
        qa[kc].q[0] = *(const ulonglong2*)p;
        qa[kc].q[1] = *(const ulonglong2*)(p + 8);
    }

    float mrun[8], lrun[8];
    v8f oacc[4];
#pragma unroll
    for (int r = 0; r < 8; ++r) { mrun[r] = -1e30f; lrun[r] = 0.0f; }
#pragma unroll
    for (int d = 0; d < 4; ++d) oacc[d] = zero_v8f();

    // uniform trip count across the block's 8 waves (barrier legality)
    int nk = Pc + qblock + 128;
    if (nk > Lc) nk = Lc;
    const int KB = (nk + 31) / 32;

    __shared__ __bf16 Plds[8][16 * 32];

    for (int kbi = 0; kbi < KB; ++kbi) {
        const int kk = kbi * 32;

        // scores for two 16-key tiles, each K=32 over dk (2 chunks)
        v8f sc[2];
#pragma unroll
        for (int t = 0; t < 2; ++t) {
            sc[t] = zero_v8f();
#pragma unroll
            for (int kc = 0; kc < 2; ++kc) {
                const __bf16* kp = Kp + (size_t)(kk + t * 16 + m0) * DKc + kc * 32 + ksec;
                v16bf bk = *(const v16bf*)kp;
                sc[t] = wmma_bf16(qa[kc].v, bk, sc[t]);
            }
        }

        // scale + prefix/causal mask + online softmax
#pragma unroll
        for (int r = 0; r < 8; ++r) {
            const int qi = qbase + r + 8 * hh;
            float s0 = sc[0][r] * 0.125f;
            float s1 = sc[1][r] * 0.125f;
            const int j0 = kk + m0;
            const int j1 = kk + 16 + m0;
            if (!((j0 < Pc) || (j0 - Pc <= qi))) s0 = -1e30f;
            if (!((j1 < Pc) || (j1 - Pc <= qi))) s1 = -1e30f;

            float mx = fmaxf(s0, s1);
#pragma unroll
            for (int off = 8; off >= 1; off >>= 1)
                mx = fmaxf(mx, __shfl_xor(mx, off));
            const float mnew = fmaxf(mrun[r], mx);
            const float alpha = __expf(mrun[r] - mnew);
            const float p0 = __expf(s0 - mnew);
            const float p1 = __expf(s1 - mnew);
            float rs = p0 + p1;
#pragma unroll
            for (int off = 8; off >= 1; off >>= 1)
                rs += __shfl_xor(rs, off);
            lrun[r] = lrun[r] * alpha + rs;
            mrun[r] = mnew;
            sc[0][r] = p0;
            sc[1][r] = p1;
#pragma unroll
            for (int d = 0; d < 4; ++d) oacc[d][r] *= alpha;
        }

        // stage P tile (C layout -> row-major 16x32 bf16 in LDS)
#pragma unroll
        for (int t = 0; t < 2; ++t)
#pragma unroll
            for (int r = 0; r < 8; ++r)
                Plds[w][(r + 8 * hh) * 32 + t * 16 + m0] = (__bf16)sc[t][r];
        __syncthreads();

        // reload P as an A-fragment (16x32)
        AFragU pa;
        const __bf16* pl = &Plds[w][m0 * 32 + khalf];
        pa.q[0] = *(const ulonglong2*)pl;
    pa.q[1] = *(const ulonglong2*)(pl + 8);

        // O += P * V  (V pre-transposed: B-frag = contiguous keys per dk lane)
#pragma unroll
        for (int d = 0; d < 4; ++d) {
            const __bf16* vp = Vp + (size_t)(d * 16 + m0) * Lp + kk + ksec;
            v16bf bv = *(const v16bf*)vp;
            oacc[d] = wmma_bf16(pa.v, bv, oacc[d]);
        }
        __syncthreads();
    }

    // epilogue: normalize rows, scatter to [B,S,D] f32
    const int bb = bh / Hc;
    const int h  = bh % Hc;
#pragma unroll
    for (int r = 0; r < 8; ++r) {
        const float inv = 1.0f / lrun[r];
        const int srow = qbase + r + 8 * hh;
#pragma unroll
        for (int d = 0; d < 4; ++d) {
            const int dk = d * 16 + m0;
            O32[((size_t)bb * Sc + srow) * Dc + h * DKc + dk] = oacc[d][r] * inv;
        }
    }
}

// ---------------- host side ----------------
extern "C" void kernel_launch(void* const* d_in, const int* in_sizes, int n_in,
                              void* d_out, int out_size, void* d_ws, size_t ws_size,
                              hipStream_t stream) {
    (void)in_sizes; (void)n_in; (void)out_size; (void)ws_size;
    const float* q  = (const float*)d_in[0];
    const float* k  = (const float*)d_in[1];
    const float* v  = (const float*)d_in[2];
    // d_in[3] = mask (tril) — computed analytically in-kernel
    const float* Wq = (const float*)d_in[4];
    const float* bq = (const float*)d_in[5];
    const float* Wk = (const float*)d_in[6];
    const float* bk = (const float*)d_in[7];
    const float* Wv = (const float*)d_in[8];
    const float* bv = (const float*)d_in[9];
    const float* Wo = (const float*)d_in[10];
    const float* bo = (const float*)d_in[11];
    const float* pk = (const float*)d_in[12];
    const float* pv = (const float*)d_in[13];

    char* ws = (char*)d_ws;
    size_t off = 0;
    auto alloc = [&](size_t bytes) { size_t o = off; off += (bytes + 255) & ~(size_t)255; return o; };
    const size_t wBytes  = (size_t)Dc * Dc * sizeof(__bf16);
    const size_t qBytes  = (size_t)Bc * Hc * Sc * DKc * sizeof(__bf16);
    const size_t kBytes  = (size_t)Bc * Hc * Lp * DKc * sizeof(__bf16);
    __bf16* Wtq = (__bf16*)(ws + alloc(wBytes));
    __bf16* Wtk = (__bf16*)(ws + alloc(wBytes));
    __bf16* Wtv = (__bf16*)(ws + alloc(wBytes));
    __bf16* Wto = (__bf16*)(ws + alloc(wBytes));
    __bf16* Qbf = (__bf16*)(ws + alloc(qBytes));
    __bf16* Kbf = (__bf16*)(ws + alloc(kBytes));
    __bf16* Vt  = (__bf16*)(ws + alloc(kBytes));
    float*  O32 = (float*)(ws + alloc((size_t)Bc * Sc * Dc * sizeof(float)));

    const dim3 blk(256);
    // weight transposes
    wt_conv_kernel<<<dim3((Dc * Dc) / 256), blk, 0, stream>>>(Wq, Wtq);
    wt_conv_kernel<<<dim3((Dc * Dc) / 256), blk, 0, stream>>>(Wk, Wtk);
    wt_conv_kernel<<<dim3((Dc * Dc) / 256), blk, 0, stream>>>(Wv, Wtv);
    wt_conv_kernel<<<dim3((Dc * Dc) / 256), blk, 0, stream>>>(Wo, Wto);
    // prefix K/V + pad
    prefix_fill_kernel<<<dim3((Bc * Hc * 32 * DKc) / 256), blk, 0, stream>>>(pk, pv, Kbf, Vt);
    // projections
    const dim3 ggrid(Dc / 128, Mrows / 128);
    gemm_kernel<1><<<ggrid, blk, 0, stream>>>(q, Wtq, bq, (void*)Qbf);
    gemm_kernel<2><<<ggrid, blk, 0, stream>>>(k, Wtk, bk, (void*)Kbf);
    gemm_kernel<3><<<ggrid, blk, 0, stream>>>(v, Wtv, bv, (void*)Vt);
    // attention
    flash_attn_kernel<<<dim3(Bc * Hc, Sc / 128), blk, 0, stream>>>(Qbf, Kbf, Vt, O32);
    // output projection -> d_out (f32)
    gemm_kernel<0><<<ggrid, blk, 0, stream>>>(O32, Wto, bo, d_out);
}
